// L2Prompt_88545045775200
// MI455X (gfx1250) — matrix operand order repaired and checked by reading, not compiled, not run on hardware
//
#include <hip/hip_runtime.h>
#include <hip/hip_bf16.h>
#include <cfloat>
#include <cmath>

#define B_DIM 8192
#define P_DIM 1024
#define D_DIM 2048
#define K_SEL 5
#define EPS_F 1e-8f
#define COEFF 0.5f

typedef __attribute__((ext_vector_type(16))) __bf16 v16bf;
typedef __attribute__((ext_vector_type(8)))  __bf16 v8bf;
typedef __attribute__((ext_vector_type(4)))  __bf16 v4bf;
typedef __attribute__((ext_vector_type(8)))  float  v8f;

// ---------------------------------------------------------------------------
// Pass 0: fp32 -> bf16 conversion + row L2 norms. One block per row.
// ---------------------------------------------------------------------------
__global__ void __launch_bounds__(256)
cvt_norm_kernel(const float* __restrict__ src, __bf16* __restrict__ dst,
                float* __restrict__ norms) {
    __shared__ float red[256];
    const int row = blockIdx.x;
    const int tid = threadIdx.x;
    const float4* s4 = (const float4*)(src + (size_t)row * D_DIM);
    __bf16* d = dst + (size_t)row * D_DIM;

    float acc = 0.0f;
    for (int i = tid; i < D_DIM / 4; i += 256) {
        float4 v = s4[i];
        acc += v.x * v.x + v.y * v.y + v.z * v.z + v.w * v.w;
        v4bf o;
        o[0] = (__bf16)v.x; o[1] = (__bf16)v.y;
        o[2] = (__bf16)v.z; o[3] = (__bf16)v.w;
        *(v4bf*)(d + 4 * i) = o;
    }
    red[tid] = acc;
    __syncthreads();
    for (int off = 128; off > 0; off >>= 1) {
        if (tid < off) red[tid] += red[tid + off];
        __syncthreads();
    }
    if (tid == 0) norms[row] = sqrtf(red[0]);
}

// ---------------------------------------------------------------------------
// Zero the two scalar output slots (harness poisons d_out).
// ---------------------------------------------------------------------------
__global__ void zero_scalars_kernel(float* __restrict__ p) {
    p[0] = 0.0f;
    p[1] = 0.0f;
}

// ---------------------------------------------------------------------------
// Pass 1: register-blocked WMMA GEMM (bf16 -> f32) + cosine->score epilogue.
// Each wave owns a 32x64 output patch: 2 A-fragments x 4 B-fragments = 8 WMMA
// tiles, so per 32-wide k-step it issues 12 b128 loads for 8 WMMAs
// (1.5 loads/WMMA vs 4 unblocked). Block = 8 waves (2x4) -> 64x256 tile.
// ---------------------------------------------------------------------------
__global__ void __launch_bounds__(256)
gemm_score_kernel(const __bf16* __restrict__ qbf, const __bf16* __restrict__ kbf,
                  const float* __restrict__ qn, const float* __restrict__ kn,
                  float* __restrict__ score) {
    const int tid  = threadIdx.x;
    const int lane = tid & 31;
    const int wave = tid >> 5;
    const int half = lane >> 4;      // lane half selects K sub-range
    const int lr   = lane & 15;      // M index (A) / N index (B, C, D)

    const int rowBase = blockIdx.y * 64  + (wave >> 2) * 32;  // 2 row-waves
    const int colBase = blockIdx.x * 256 + (wave & 3) * 64;   // 4 col-waves

    // A fragments (16x32 bf16): lane = row M=lr; VGPR0..3 = K kOff..kOff+7,
    // VGPR4..7 = K kOff+8..kOff+15, kOff = half*16 (contiguous, matches B).
    const __bf16* aRow0 = qbf + (size_t)(rowBase + lr) * D_DIM + half * 16;
    const __bf16* aRow1 = aRow0 + (size_t)16 * D_DIM;
    // B fragments (32x16 bf16): lane = column N=lr; lane-half 0 -> K 0..15,
    // lane-half 1 -> K 16..31, contiguous (sparse-B layout pattern).
    const __bf16* bRow0 = kbf + (size_t)(colBase + lr) * D_DIM + half * 16;
    const __bf16* bRow1 = bRow0 + (size_t)16 * D_DIM;
    const __bf16* bRow2 = bRow0 + (size_t)32 * D_DIM;
    const __bf16* bRow3 = bRow0 + (size_t)48 * D_DIM;

    v8f acc00 = {}, acc01 = {}, acc02 = {}, acc03 = {};
    v8f acc10 = {}, acc11 = {}, acc12 = {}, acc13 = {};

    for (int kk = 0; kk < D_DIM; kk += 32) {
        union { v16bf v; v8bf h[2]; } a0, a1;
        a0.h[0] = *(const v8bf*)(aRow0 + kk);
        a0.h[1] = *(const v8bf*)(aRow0 + kk + 8);
        a1.h[0] = *(const v8bf*)(aRow1 + kk);
        a1.h[1] = *(const v8bf*)(aRow1 + kk + 8);
        const v16bf b0 = *(const v16bf*)(bRow0 + kk);
        const v16bf b1 = *(const v16bf*)(bRow1 + kk);
        const v16bf b2 = *(const v16bf*)(bRow2 + kk);
        const v16bf b3 = *(const v16bf*)(bRow3 + kk);

        acc00 = __builtin_amdgcn_wmma_f32_16x16x32_bf16(false, a0.v, false, b0,
                                                        (short)0, acc00, false, false);
        acc01 = __builtin_amdgcn_wmma_f32_16x16x32_bf16(false, a0.v, false, b1,
                                                        (short)0, acc01, false, false);
        acc02 = __builtin_amdgcn_wmma_f32_16x16x32_bf16(false, a0.v, false, b2,
                                                        (short)0, acc02, false, false);
        acc03 = __builtin_amdgcn_wmma_f32_16x16x32_bf16(false, a0.v, false, b3,
                                                        (short)0, acc03, false, false);
        acc10 = __builtin_amdgcn_wmma_f32_16x16x32_bf16(false, a1.v, false, b0,
                                                        (short)0, acc10, false, false);
        acc11 = __builtin_amdgcn_wmma_f32_16x16x32_bf16(false, a1.v, false, b1,
                                                        (short)0, acc11, false, false);
        acc12 = __builtin_amdgcn_wmma_f32_16x16x32_bf16(false, a1.v, false, b2,
                                                        (short)0, acc12, false, false);
        acc13 = __builtin_amdgcn_wmma_f32_16x16x32_bf16(false, a1.v, false, b3,
                                                        (short)0, acc13, false, false);
    }

    // Epilogue. C/D layout: lane lr = N (within tile), VGPR r -> M = r + 8*half.
    v8f accs[2][4] = {{acc00, acc01, acc02, acc03}, {acc10, acc11, acc12, acc13}};
#pragma unroll
    for (int j = 0; j < 4; ++j) {
        const int   n   = colBase + j * 16 + lr;
        const float knv = kn[n];
#pragma unroll
        for (int i = 0; i < 2; ++i) {
            const int mBase = rowBase + i * 16 + half * 8;
#pragma unroll
            for (int r = 0; r < 8; ++r) {
                const int   m     = mBase + r;
                const float denom = fmaxf(qn[m] * knv, EPS_F);
                score[(size_t)m * P_DIM + n] = 1.0f - accs[i][j][r] / denom;
            }
        }
    }
}

// ---------------------------------------------------------------------------
// Pass 2: per-row softmax entropy + top-5 smallest scores. One block per row.
// scalars[0] += sum of 5 smallest scores; scalars[1] += row entropy.
// ---------------------------------------------------------------------------
__global__ void __launch_bounds__(256)
row_topk_entropy_kernel(const float* __restrict__ score, int* __restrict__ idxOut,
                        float* __restrict__ scalars) {
    __shared__ float s[P_DIM];
    __shared__ float rv[256];
    __shared__ float rv2[256];
    __shared__ int   ri[256];

    const int row = blockIdx.x;
    const int tid = threadIdx.x;
    const float* sr = score + (size_t)row * P_DIM;

    for (int i = tid; i < P_DIM; i += 256) s[i] = sr[i];
    __syncthreads();

    // --- row max ---
    float m = -FLT_MAX;
    for (int i = tid; i < P_DIM; i += 256) m = fmaxf(m, s[i]);
    rv[tid] = m;
    __syncthreads();
    for (int off = 128; off > 0; off >>= 1) {
        if (tid < off) rv[tid] = fmaxf(rv[tid], rv[tid + off]);
        __syncthreads();
    }
    m = rv[0];
    __syncthreads();

    // --- sum exp(s-m) and sum s*exp(s-m) ---
    float se = 0.0f, sse = 0.0f;
    for (int i = tid; i < P_DIM; i += 256) {
        float e = expf(s[i] - m);
        se  += e;
        sse += s[i] * e;
    }
    rv[tid] = se; rv2[tid] = sse;
    __syncthreads();
    for (int off = 128; off > 0; off >>= 1) {
        if (tid < off) { rv[tid] += rv[tid + off]; rv2[tid] += rv2[tid + off]; }
        __syncthreads();
    }
    if (tid == 0) {
        const float Z = rv[0];
        const float H = (m + logf(Z)) - rv2[0] / Z;   // logZ - E[s]
        atomicAdd(&scalars[1], H);
    }
    __syncthreads();

    // --- top-5 smallest via 5 argmin rounds (low-index tie-break) ---
    for (int t = 0; t < K_SEL; ++t) {
        float bv = FLT_MAX;
        int   bi = P_DIM;
        for (int i = tid; i < P_DIM; i += 256) {
            float v = s[i];
            if (v < bv) { bv = v; bi = i; }
        }
        rv[tid] = bv; ri[tid] = bi;
        __syncthreads();
        for (int off = 128; off > 0; off >>= 1) {
            if (tid < off) {
                const float ov = rv[tid + off];
                const int   oi = ri[tid + off];
                if (ov < rv[tid] || (ov == rv[tid] && oi < ri[tid])) {
                    rv[tid] = ov; ri[tid] = oi;
                }
            }
            __syncthreads();
        }
        if (tid == 0) {
            idxOut[row * K_SEL + t] = ri[0];
            rv2[t] = rv[0];          // stash selected value
            s[ri[0]] = FLT_MAX;      // remove from next round
        }
        __syncthreads();
    }
    if (tid == 0)
        atomicAdd(&scalars[0], rv2[0] + rv2[1] + rv2[2] + rv2[3] + rv2[4]);
}

// ---------------------------------------------------------------------------
// Pass 3: prompted = ppg + 0.5 * sum_{j<5} prompt[idx[b][j]].  prompt (8 MB)
// is L2-resident, so the 5-way gather is cheap. One block per row, float4.
// ---------------------------------------------------------------------------
__global__ void __launch_bounds__(256)
gather_add_kernel(const float* __restrict__ ppg, const float* __restrict__ prompt,
                  const int* __restrict__ idx, float* __restrict__ out) {
    const int row = blockIdx.x;
    const int* id = idx + row * K_SEL;
    const int i0 = id[0], i1 = id[1], i2 = id[2], i3 = id[3], i4 = id[4];

    const float4* src = (const float4*)(ppg + (size_t)row * D_DIM);
    const float4* p0  = (const float4*)(prompt + (size_t)i0 * D_DIM);
    const float4* p1  = (const float4*)(prompt + (size_t)i1 * D_DIM);
    const float4* p2  = (const float4*)(prompt + (size_t)i2 * D_DIM);
    const float4* p3  = (const float4*)(prompt + (size_t)i3 * D_DIM);
    const float4* p4  = (const float4*)(prompt + (size_t)i4 * D_DIM);
    float4* dst = (float4*)(out + (size_t)row * D_DIM);

    for (int i = threadIdx.x; i < D_DIM / 4; i += 256) {
        const float4 a = src[i];
        const float4 b0 = p0[i], b1 = p1[i], b2 = p2[i], b3 = p3[i], b4 = p4[i];
        float4 o;
        o.x = a.x + COEFF * (b0.x + b1.x + b2.x + b3.x + b4.x);
        o.y = a.y + COEFF * (b0.y + b1.y + b2.y + b3.y + b4.y);
        o.z = a.z + COEFF * (b0.z + b1.z + b2.z + b3.z + b4.z);
        o.w = a.w + COEFF * (b0.w + b1.w + b2.w + b3.w + b4.w);
        dst[i] = o;
    }
}

// ---------------------------------------------------------------------------
extern "C" void kernel_launch(void* const* d_in, const int* in_sizes, int n_in,
                              void* d_out, int out_size, void* d_ws, size_t ws_size,
                              hipStream_t stream) {
    (void)in_sizes; (void)n_in; (void)out_size; (void)ws_size;

    const float* ppg    = (const float*)d_in[0];
    const float* keys   = (const float*)d_in[1];
    const float* prompt = (const float*)d_in[2];
    float*       out    = (float*)d_out;

    // workspace carve-up (all offsets 256B-aligned by construction)
    char*   ws    = (char*)d_ws;
    size_t  off   = 0;
    __bf16* qbf   = (__bf16*)(ws + off); off += (size_t)B_DIM * D_DIM * sizeof(__bf16);
    __bf16* kbf   = (__bf16*)(ws + off); off += (size_t)P_DIM * D_DIM * sizeof(__bf16);
    float*  qn    = (float*)(ws + off);  off += (size_t)B_DIM * sizeof(float);
    float*  kn    = (float*)(ws + off);  off += (size_t)P_DIM * sizeof(float);
    float*  score = (float*)(ws + off);  off += (size_t)B_DIM * P_DIM * sizeof(float);
    int*    idx   = (int*)(ws + off);

    float* scalars = out + (size_t)B_DIM * D_DIM;   // [sel_score_sum, entropy]

    cvt_norm_kernel<<<B_DIM, 256, 0, stream>>>(ppg,  qbf, qn);
    cvt_norm_kernel<<<P_DIM, 256, 0, stream>>>(keys, kbf, kn);
    zero_scalars_kernel<<<1, 1, 0, stream>>>(scalars);

    dim3 gemmGrid(P_DIM / 256, B_DIM / 64);   // (4, 128)
    gemm_score_kernel<<<gemmGrid, 256, 0, stream>>>(qbf, kbf, qn, kn, score);

    row_topk_entropy_kernel<<<B_DIM, 256, 0, stream>>>(score, idx, scalars);
    gather_add_kernel<<<B_DIM, 256, 0, stream>>>(ppg, prompt, idx, out);
}
